// ClusteringLayer_1297080124196
// MI455X (gfx1250) — matrix operand order repaired and checked by reading, not compiled, not run on hardware
//
#include <hip/hip_runtime.h>
#include <hip/hip_bf16.h>

// ---------------------------------------------------------------------------
// Soft clustering assignments (Student-t kernel + row normalization).
//   cross = x @ centroids^T  via bf16 split (hi/lo) WMMA, fp32 accumulate.
// MI455X / gfx1250: wave32, v_wmma_f32_16x16x32_bf16, LDS-staged A operand.
// ---------------------------------------------------------------------------

typedef __attribute__((ext_vector_type(16))) __bf16 v16bf;
typedef __attribute__((ext_vector_type(2)))  __bf16 v2bf;
typedef __attribute__((ext_vector_type(8)))  float  v8f;
typedef __attribute__((ext_vector_type(4)))  float  v4f;

#define NF   256          // features (K)
#define NC   256          // clusters (N)
#define MBLK 32           // rows per workgroup

static __device__ __forceinline__ v8f wmma_bf16(v16bf a, v16bf b, v8f c) {
  // (neg_a, A, neg_b, B, c_mod, C, reuse_a, reuse_b)
  return __builtin_amdgcn_wmma_f32_16x16x32_bf16(false, a, false, b,
                                                 (short)0, c, false, false);
}

// ---------------------------------------------------------------------------
// Prep: split centroids into bf16 hi/lo, pre-swizzled into the WMMA B-operand
// register layout (contiguous K order, not the ISA table order; B 32x16 per 32-K chunk):
//   lanes 0-15  : column N = lane,      elems 0..15 = K 0..15 of chunk
//   lanes 16-31 : column N = lane - 16, elems 0..15 = K 16..31 of chunk
// Linear index: ((tileN*8 + kChunk)*32 + lane)*16 + elem
// Also computes c_sq[n] = sum_k centroids[n][k]^2.
// ---------------------------------------------------------------------------
__global__ void prep_centroids_kernel(const float* __restrict__ cent,
                                      __bf16* __restrict__ bhi,
                                      __bf16* __restrict__ blo,
                                      float* __restrict__ csq) {
  const int n = blockIdx.x;     // centroid row 0..255
  const int k = threadIdx.x;    // feature 0..255

  float v = cent[n * NF + k];
  __bf16 h = (__bf16)v;
  __bf16 l = (__bf16)(v - (float)h);

  const int tileN = n >> 4, nn = n & 15;
  const int kc = k >> 5, kk = k & 31;
  const int lane = nn + ((kk >= 16) ? 16 : 0);
  const int elem = kk & 15;
  const int idx = ((tileN * 8 + kc) * 32 + lane) * 16 + elem;
  bhi[idx] = h;
  blo[idx] = l;

  // block reduction of v*v -> csq[n]
  float s = v * v;
#pragma unroll
  for (int m = 1; m < 32; m <<= 1) s += __shfl_xor(s, m, 32);
  __shared__ float part[8];
  if ((threadIdx.x & 31) == 0) part[threadIdx.x >> 5] = s;
  __syncthreads();
  if (threadIdx.x < 8) {
    float t = part[threadIdx.x];
    t += __shfl_xor(t, 1, 32);
    t += __shfl_xor(t, 2, 32);
    t += __shfl_xor(t, 4, 32);
    if (threadIdx.x == 0) csq[n] = t;
  }
}

// ---------------------------------------------------------------------------
// Main kernel: one block = 32 rows x 256 cols. 8 waves:
//   wave = m_tile(2) x n_group(4); each wave -> 16 rows x 64 cols (4 accums).
//
// Stage 1 (cooperative, once per block): load the 32x256 fp32 x-tile, split
// into bf16 hi/lo ONCE, store into LDS pre-swizzled in the WMMA A-operand
// lane layout (contiguous K order matching B, not the ISA table order; A 16x32):
//   lane<16 : M=lane,    elems 0..15 -> K 0..15
//   lane>=16: M=lane-16, elems 0..15 -> K 16..31
// x_sq per row falls out of the same pass (8-lane shfl reduce).
//
// Stage 2: K-loop of 8 steps of 32; A fragments via contiguous ds_load,
// B fragments via contiguous global loads (L2-resident); 3 WMMA per tile.
//
// Epilogue: q = 1/(1 + relu(x2 - 2*cross + c2)) -> LDS -> row sums ->
// coalesced normalized store.
// ---------------------------------------------------------------------------
__global__ __launch_bounds__(256)
void cluster_assign_kernel(const float* __restrict__ x,
                           const __bf16* __restrict__ bhi,
                           const __bf16* __restrict__ blo,
                           const float* __restrict__ csq,
                           float* __restrict__ out) {
  __shared__ __bf16 a_hi_s[2 * 8 * 32 * 16];   // [m_tile][kc][lane][elem]
  __shared__ __bf16 a_lo_s[2 * 8 * 32 * 16];
  __shared__ float  q_lds[MBLK][NC + 1];       // +1 pad
  __shared__ float  xsq_lds[MBLK];
  __shared__ float  rinv_lds[MBLK];

  const int tid     = threadIdx.x;
  const int wave    = tid >> 5;       // 0..7
  const int lane    = tid & 31;
  const int m_tile  = wave >> 2;      // 0..1
  const int n_group = wave & 3;       // 0..3
  const int lrow    = lane & 15;
  const bool hiHalf = lane >= 16;

  const size_t m_base = (size_t)blockIdx.x * MBLK;   // global row base
  const int    m_wave = m_tile * 16;                 // local row base of wave

  // ---------------- Stage 1: cooperative A staging + x_sq -----------------
  {
    const int sm  = tid >> 3;   // local row 0..31
    const int skc = tid & 7;    // k-chunk 0..7
    const float* srow = x + (m_base + sm) * NF + skc * 32;

    float vals[32];
#pragma unroll
    for (int i = 0; i < 8; ++i) {
      v4f f = *(const v4f*)(srow + 4 * i);
      vals[4 * i + 0] = f[0];
      vals[4 * i + 1] = f[1];
      vals[4 * i + 2] = f[2];
      vals[4 * i + 3] = f[3];
    }

    const int mt = sm >> 4, mm = sm & 15;
    float s = 0.f;
#pragma unroll
    for (int j = 0; j < 16; ++j) {
      const int kk   = 2 * j;
      const int grp  = kk >> 4;                      // 0..1
      const int ln   = mm + (grp << 4);              // target lane
      const int elem = kk & 15;                      // even -> 4B aligned
      const int base = ((mt * 8 + skc) * 32 + ln) * 16 + elem;

      const float f0 = vals[kk], f1 = vals[kk + 1];
      const __bf16 h0 = (__bf16)f0, h1 = (__bf16)f1;
      const __bf16 l0 = (__bf16)(f0 - (float)h0);
      const __bf16 l1 = (__bf16)(f1 - (float)h1);
      *(v2bf*)(a_hi_s + base) = (v2bf){h0, h1};
      *(v2bf*)(a_lo_s + base) = (v2bf){l0, l1};
      s = fmaf(f0, f0, s);
      s = fmaf(f1, f1, s);
    }
    // 8 consecutive lanes hold complementary K chunks of row sm
    s += __shfl_xor(s, 1, 32);
    s += __shfl_xor(s, 2, 32);
    s += __shfl_xor(s, 4, 32);
    if (skc == 0) xsq_lds[sm] = s;
  }
  __syncthreads();

  // ---------------- Stage 2: WMMA K-loop ----------------------------------
  v8f acc[4];
#pragma unroll
  for (int t = 0; t < 4; ++t) acc[t] = (v8f){};

  const int tn0 = n_group * 4;
  for (int ks = 0; ks < 8; ++ks) {
    const int aoff = ((m_tile * 8 + ks) * 32 + lane) * 16;
    v16bf ahi = *(const v16bf*)(a_hi_s + aoff);
    v16bf alo = *(const v16bf*)(a_lo_s + aoff);

#pragma unroll
    for (int t = 0; t < 4; ++t) {
      const size_t boff = ((size_t)((tn0 + t) * 8 + ks) * 32 + lane) * 16;
      v16bf bh = *(const v16bf*)(bhi + boff);
      v16bf bl = *(const v16bf*)(blo + boff);
      acc[t] = wmma_bf16(ahi, bh, acc[t]);   // hi*hi
      acc[t] = wmma_bf16(ahi, bl, acc[t]);   // hi*lo
      acc[t] = wmma_bf16(alo, bh, acc[t]);   // lo*hi
    }
  }

  // ---------------- Epilogue ----------------------------------------------
  // c_sq for this lane's 4 column tiles
  float cs[4];
#pragma unroll
  for (int t = 0; t < 4; ++t) cs[t] = csq[n_group * 64 + t * 16 + lrow];

  // C/D layout (ISA 7.12.2): lanes 0-15 N=lane, VGPR j -> M=j;
  //                          lanes 16-31 N=lane-16, VGPR j -> M=8+j.
#pragma unroll
  for (int j = 0; j < 8; ++j) {
    const int m_loc = m_wave + j + (hiHalf ? 8 : 0);
    const float xs = xsq_lds[m_loc];
#pragma unroll
    for (int t = 0; t < 4; ++t) {
      const int n = n_group * 64 + t * 16 + lrow;
      float d = fmaf(-2.f, acc[t][j], xs) + cs[t];
      d = fmaxf(d, 0.f);
      q_lds[m_loc][n] = 1.f / (1.f + d);
    }
  }
  __syncthreads();

  // Row sums: thread -> (row = tid/8, segment = tid%8 of 32 elems)
  {
    const int r = tid >> 3, seg = tid & 7;
    float p = 0.f;
#pragma unroll
    for (int i = 0; i < 32; ++i) p += q_lds[r][seg * 32 + i];
    p += __shfl_xor(p, 1, 32);
    p += __shfl_xor(p, 2, 32);
    p += __shfl_xor(p, 4, 32);
    if (seg == 0) rinv_lds[r] = 1.f / p;
  }
  __syncthreads();

  // Coalesced normalized store: 32 rows x 256 cols
#pragma unroll 4
  for (int r = 0; r < MBLK; ++r) {
    out[(m_base + r) * NC + tid] = q_lds[r][tid] * rinv_lds[r];
  }
}

// ---------------------------------------------------------------------------
extern "C" void kernel_launch(void* const* d_in, const int* in_sizes, int n_in,
                              void* d_out, int out_size, void* d_ws, size_t ws_size,
                              hipStream_t stream) {
  const float* x    = (const float*)d_in[0];
  const float* cent = (const float*)d_in[1];
  float* out        = (float*)d_out;

  const int n_samples = in_sizes[0] / NF;   // 262144

  __bf16* bhi = (__bf16*)d_ws;
  __bf16* blo = bhi + (size_t)NC * NF;
  float*  csq = (float*)(blo + (size_t)NC * NF);

  prep_centroids_kernel<<<NC, NF, 0, stream>>>(cent, bhi, blo, csq);
  cluster_assign_kernel<<<n_samples / MBLK, 256, 0, stream>>>(x, bhi, blo, csq, out);
}
